// net_rd_42984032699194
// MI455X (gfx1250) — matrix operand order repaired and checked by reading, not compiled, not run on hardware
//
#include <hip/hip_runtime.h>

#define DS 256
#define NEXP 16
#define NBATCH 32768
#define KTOT 4096   // NEXP * DS

typedef __attribute__((ext_vector_type(16))) __bf16 v16bf;
typedef __attribute__((ext_vector_type(8)))  float  v8f;
typedef __attribute__((ext_vector_type(4)))  unsigned v4u;
typedef __attribute__((ext_vector_type(8)))  int    v8i;
typedef __attribute__((ext_vector_type(4)))  int    v4i;

struct __attribute__((aligned(16))) U4 { unsigned x, y, z, w; };
struct __attribute__((aligned(16))) F4 { float x, y, z, w; };
struct __attribute__((aligned(8)))  U2 { unsigned x, y; };

union FragU { v16bf v; U4 q[2]; };

#if defined(__HIP_DEVICE_COMPILE__) && \
    __has_builtin(__builtin_amdgcn_tensor_load_to_lds) && \
    __has_builtin(__builtin_amdgcn_s_wait_tensorcnt)
#define HAVE_TDM 1
#else
#define HAVE_TDM 0
#endif

__device__ __forceinline__ unsigned short f2bf(float f) {
    union { float f; unsigned u; } c; c.f = f;
    unsigned r = 0x7FFFu + ((c.u >> 16) & 1u);
    return (unsigned short)((c.u + r) >> 16);
}
__device__ __forceinline__ float bf2f(unsigned short u) {
    union { unsigned u; float f; } c; c.u = ((unsigned)u) << 16; return c.f;
}
__device__ __forceinline__ unsigned pack2(float a, float b) {
    return (unsigned)f2bf(a) | ((unsigned)f2bf(b) << 16);
}

// ---------------------------------------------------------------------------
// diff_bf16[b,j] = bf16(x_tar[j] - x[b,j])
// ---------------------------------------------------------------------------
__global__ __launch_bounds__(256) void prep_diff_kernel(
    const float* __restrict__ x, const float* __restrict__ xtar,
    unsigned short* __restrict__ diff_bf)
{
    int idx = (blockIdx.x * 256 + threadIdx.x) * 4;   // over B*DS elements
    int j = idx & (DS - 1);
    F4 xv = *(const F4*)(x + idx);
    F4 tv = *(const F4*)(xtar + j);
    U2 o;
    o.x = pack2(tv.x - xv.x, tv.y - xv.y);
    o.y = pack2(tv.z - xv.z, tv.w - xv.w);
    *(U2*)(diff_bf + idx) = o;
}

// ---------------------------------------------------------------------------
// Afat[i, n*256 + j] = bf16(B_A[n,i,j] + C_A[n,i,j])   (K-major rows of 4096)
// ---------------------------------------------------------------------------
__global__ __launch_bounds__(256) void prep_afat_kernel(
    const float* __restrict__ BA, const float* __restrict__ CA,
    unsigned short* __restrict__ afat)
{
    size_t t = ((size_t)blockIdx.x * 256 + threadIdx.x) * 4;  // over 256*4096
    int i   = (int)(t >> 12);
    int rem = (int)t & 4095;
    int n = rem >> 8;
    int j = rem & 255;
    size_t src = ((size_t)n << 16) + ((size_t)i << 8) + (size_t)j;
    F4 bv = *(const F4*)(BA + src);
    F4 cv = *(const F4*)(CA + src);
    U2 o;
    o.x = pack2(bv.x + cv.x, bv.y + cv.y);
    o.y = pack2(bv.z + cv.z, bv.w + cv.w);
    *(U2*)(afat + t) = o;
}

// ---------------------------------------------------------------------------
// w1bf = bf16(W1); c1[o] = dot(W1[o,:], x_tar) + b1[o]
// ---------------------------------------------------------------------------
__global__ __launch_bounds__(128) void prep_w1c1_kernel(
    const float* __restrict__ W1, const float* __restrict__ b1,
    const float* __restrict__ xtar,
    unsigned short* __restrict__ w1bf, float* __restrict__ c1)
{
    __shared__ float sx[DS];
    int tid = threadIdx.x;                 // 128 threads, one per output o
    sx[tid] = xtar[tid];
    sx[tid + 128] = xtar[tid + 128];
    __syncthreads();
    const float* wrow = W1 + tid * DS;
    unsigned short* orow = w1bf + tid * DS;
    float acc = b1[tid];
    for (int jj = 0; jj < DS; jj += 4) {
        F4 wv = *(const F4*)(wrow + jj);
        acc += wv.x * sx[jj] + wv.y * sx[jj + 1] + wv.z * sx[jj + 2] + wv.w * sx[jj + 3];
        U2 o; o.x = pack2(wv.x, wv.y); o.y = pack2(wv.z, wv.w);
        *(U2*)(orow + jj) = o;
    }
    c1[tid] = acc;
}

// ---------------------------------------------------------------------------
// h1_bf16 = bf16(relu(c1 - diff @ W1bf^T))     [B,128], WMMA bf16
// Block: 256 thr = 8 waves (4M x 2N), wave tile 64x64, K=256.
// ---------------------------------------------------------------------------
__global__ __launch_bounds__(256, 1) void gemm_h1_kernel(
    const unsigned short* __restrict__ diff_bf,  // [B,256]
    const unsigned short* __restrict__ w1bf,     // [128,256]
    const float* __restrict__ c1,                // [128]
    unsigned short* __restrict__ h1bf)           // [B,128]
{
    const int tid  = threadIdx.x;
    const int lane = tid & 31;
    const int wave = tid >> 5;
    const int wm = wave >> 1;          // 0..3
    const int wn = wave & 1;           // 0..1
    const int mblock = blockIdx.x * 256;
    const int arow = lane & 15;
    const int ksel = lane >> 4;        // 0/1 half-wave K select
    const int m0 = mblock + wm * 64;
    const int n0 = wn * 64;

    v8f acc[4][4];
    const v8f vzero = {0.f,0.f,0.f,0.f,0.f,0.f,0.f,0.f};
#pragma unroll
    for (int mi = 0; mi < 4; ++mi)
#pragma unroll
        for (int ni = 0; ni < 4; ++ni) acc[mi][ni] = vzero;

    // Hoisted loop-invariant row bases; K offsets become immediates below.
    const unsigned short* ap[4];
    const unsigned short* bp[4];
#pragma unroll
    for (int mi = 0; mi < 4; ++mi)
        ap[mi] = diff_bf + (size_t)(m0 + mi * 16 + arow) * DS + ksel * 8;
#pragma unroll
    for (int ni = 0; ni < 4; ++ni)
        bp[ni] = w1bf + (size_t)(n0 + ni * 16 + arow) * DS + ksel * 8;

#pragma unroll
    for (int kk = 0; kk < 8; ++kk) {
        const int k0 = kk * 32;
        FragU bfr[4];
#pragma unroll
        for (int ni = 0; ni < 4; ++ni) {
            bfr[ni].q[0] = *(const U4*)(bp[ni] + k0);
            bfr[ni].q[1] = *(const U4*)(bp[ni] + k0 + 16);
        }
#pragma unroll
        for (int mi = 0; mi < 4; ++mi) {
            FragU a;
            a.q[0] = *(const U4*)(ap[mi] + k0);
            a.q[1] = *(const U4*)(ap[mi] + k0 + 16);
#pragma unroll
            for (int ni = 0; ni < 4; ++ni)
                acc[mi][ni] = __builtin_amdgcn_wmma_f32_16x16x32_bf16(
                    false, a.v, false, bfr[ni].v, (short)0, acc[mi][ni], false, false);
        }
    }

#pragma unroll
    for (int ni = 0; ni < 4; ++ni) {
        const int gcol = n0 + ni * 16 + arow;
        const float cv = c1[gcol];
#pragma unroll
        for (int mi = 0; mi < 4; ++mi)
#pragma unroll
            for (int r = 0; r < 8; ++r) {
                float v = cv - acc[mi][ni][r];
                v = fmaxf(v, 0.0f);
                h1bf[(size_t)(m0 + mi * 16 + r + ksel * 8) * 128 + gcol] = f2bf(v);
            }
    }
}

// ---------------------------------------------------------------------------
// Gating tail: layers 2..4 + softmax, thread-per-row, weights in LDS (43KB).
// ---------------------------------------------------------------------------
__global__ __launch_bounds__(256) void gating_tail_kernel(
    const unsigned short* __restrict__ h1bf,
    const float* __restrict__ W2, const float* __restrict__ b2,
    const float* __restrict__ W3, const float* __restrict__ b3,
    const float* __restrict__ W4, const float* __restrict__ b4,
    float* __restrict__ wgate)
{
    __shared__ float sW2[64 * 128];
    __shared__ float sW3[32 * 64];
    __shared__ float sW4[16 * 32];
    int tid = threadIdx.x;
    for (int i = tid; i < 64 * 128; i += 256) sW2[i] = W2[i];
    for (int i = tid; i < 32 * 64;  i += 256) sW3[i] = W3[i];
    for (int i = tid; i < 16 * 32;  i += 256) sW4[i] = W4[i];
    __syncthreads();

    int row = blockIdx.x * 256 + tid;
    const unsigned short* hrow = h1bf + (size_t)row * 128;

    float a2[64];
#pragma unroll
    for (int o = 0; o < 64; ++o) a2[o] = b2[o];
    for (int c = 0; c < 128; c += 8) {
        union { U4 q; unsigned short s[8]; } hv;
        hv.q = *(const U4*)(hrow + c);
        float h[8];
#pragma unroll
        for (int e = 0; e < 8; ++e) h[e] = bf2f(hv.s[e]);
#pragma unroll
        for (int o = 0; o < 64; ++o) {
            const float* wr = sW2 + o * 128 + c;
#pragma unroll
            for (int e = 0; e < 8; ++e) a2[o] += wr[e] * h[e];
        }
    }
#pragma unroll
    for (int o = 0; o < 64; ++o) a2[o] = fmaxf(a2[o], 0.0f);

    float a3[32];
#pragma unroll
    for (int o = 0; o < 32; ++o) {
        float s = b3[o];
        const float* wr = sW3 + o * 64;
#pragma unroll
        for (int j = 0; j < 64; ++j) s += wr[j] * a2[j];
        a3[o] = fmaxf(s, 0.0f);
    }
    float a4[16];
#pragma unroll
    for (int o = 0; o < 16; ++o) {
        float s = b4[o];
        const float* wr = sW4 + o * 32;
#pragma unroll
        for (int j = 0; j < 32; ++j) s += wr[j] * a3[j];
        a4[o] = s;
    }
    float m = a4[0];
#pragma unroll
    for (int i = 1; i < 16; ++i) m = fmaxf(m, a4[i]);
    float ssum = 0.0f;
#pragma unroll
    for (int i = 0; i < 16; ++i) { a4[i] = __expf(a4[i] - m); ssum += a4[i]; }
    float inv = 1.0f / ssum;
    float* orow = wgate + (size_t)row * 16;
#pragma unroll
    for (int i = 0; i < 16; ++i) orow[i] = a4[i] * inv;
}

// ---------------------------------------------------------------------------
// Main GEMM: out[b,i] = sum_n w[b,n] * (diff[b,:] @ Afat[i, n*256:(n+1)*256])
// Block 256 thr = 8 waves (4M x 2N), wave tile 64x32, BM=256 BN=64, K=4096.
// diff tile staged into LDS by the Tensor Data Mover (one DMA per block,
// with D# padding of 4 DWORDs every 128 DWORDs => 264-ushort padded rows).
// Per-expert accumulators folded with w[b,n] every 8 K-steps.
// ---------------------------------------------------------------------------
__global__ __launch_bounds__(256, 1) void moe_main_gemm_kernel(
    const unsigned short* __restrict__ diff_bf,  // [B,256] bf16
    const unsigned short* __restrict__ afat,     // [256,4096] bf16 K-major
    const float* __restrict__ wgate,             // [B,16]
    float* __restrict__ out)                     // [B,256]
{
    __shared__ __attribute__((aligned(16))) unsigned short s_diff[256 * 264]; // padded rows
    __shared__ float s_w[256 * 16];

    const int tid  = threadIdx.x;
    const int lane = tid & 31;
    const int wave = tid >> 5;
    const int wm = wave >> 1;              // 0..3
    const int wn = wave & 1;               // 0..1
    const int mblock = blockIdx.x * 256;
    const int nblock = blockIdx.y * 64;
    const int arow = lane & 15;
    const int ksel = lane >> 4;

    // Stage 256x16 gate weights (all threads).
    {
        const F4* src = (const F4*)(wgate + (size_t)mblock * 16);
        F4* dst = (F4*)s_w;
#pragma unroll
        for (int c = 0; c < 4; ++c)
            dst[c * 256 + tid] = src[c * 256 + tid];
    }

#if HAVE_TDM
    // Tensor Data Mover: wave 0 DMAs the whole 256x256 bf16 tile into LDS.
    // D# group0: count=1 | lds_addr | 57b global addr | type=2.
    // D# group1: data_size=2B, pad_enable, pad_interval=128 DW, pad_amount=4 DW,
    //            tensor_dim0=256, tensor_dim1=256, tile_dim0=256, tile_dim1=256,
    //            tensor_dim0_stride=256.
    if (wave == 0) {
        const unsigned lds_off = (unsigned)(size_t)(void*)s_diff;
        const unsigned long long ga =
            (unsigned long long)(size_t)(diff_bf + (size_t)mblock * DS);
        v4u g0 = { 1u,
                   lds_off,
                   (unsigned)(ga & 0xFFFFFFFFu),
                   (unsigned)((ga >> 32) & 0x01FFFFFFu) | (2u << 30) };
        v8i g1 = { (int)((1u << 16) | (1u << 20) | (6u << 22) | (3u << 25)),
                   (int)(256u << 16),   // tensor_dim0[15:0]  @ bits 63:48
                   (int)(256u << 16),   // tensor_dim1[15:0]  @ bits 111:80 (dim0 hi=0)
                   (int)(256u << 16),   // tile_dim0          @ bits 127:112
                   256,                 // tile_dim1          @ bits 143:128
                   256,                 // tensor_dim0_stride @ bits 191:160
                   0, 0 };
        v4i gz = { 0, 0, 0, 0 };
#if __clang_major__ >= 23
        v8i gz8 = { 0, 0, 0, 0, 0, 0, 0, 0 };
        __builtin_amdgcn_tensor_load_to_lds(g0, g1, gz, gz, gz8, 0);
#else
        __builtin_amdgcn_tensor_load_to_lds(g0, g1, gz, gz, 0);
#endif
        __builtin_amdgcn_s_wait_tensorcnt((short)0);
    }
#else
    // Fallback: manual copy, one padded row per thread.
    {
        const unsigned short* src = diff_bf + (size_t)(mblock + tid) * DS;
        unsigned short* dst = s_diff + tid * 264;
#pragma unroll 4
        for (int c = 0; c < DS; c += 8)
            *(U4*)(dst + c) = *(const U4*)(src + c);
    }
#endif
    __syncthreads();

    const int mrow = wm * 64;
    const int ncol = nblock + wn * 32;
    const unsigned short* bptr0 = afat + (size_t)(ncol + arow) * KTOT + ksel * 8;
    const unsigned short* bptr1 = afat + (size_t)(ncol + 16 + arow) * KTOT + ksel * 8;
    const unsigned short* arows[4];
#pragma unroll
    for (int mi = 0; mi < 4; ++mi)
        arows[mi] = s_diff + (mrow + mi * 16 + arow) * 264 + ksel * 8;

    const v8f vzero = {0.f,0.f,0.f,0.f,0.f,0.f,0.f,0.f};
    v8f outa[4][2];
#pragma unroll
    for (int mi = 0; mi < 4; ++mi) { outa[mi][0] = vzero; outa[mi][1] = vzero; }

    for (int n = 0; n < NEXP; ++n) {
        v8f acc[4][2];
#pragma unroll
        for (int mi = 0; mi < 4; ++mi) { acc[mi][0] = vzero; acc[mi][1] = vzero; }

#pragma unroll
        for (int kk = 0; kk < 8; ++kk) {
            const int k0 = n * 256 + kk * 32;
            FragU fb0, fb1;
            fb0.q[0] = *(const U4*)(bptr0 + k0);
            fb0.q[1] = *(const U4*)(bptr0 + k0 + 16);
            fb1.q[0] = *(const U4*)(bptr1 + k0);
            fb1.q[1] = *(const U4*)(bptr1 + k0 + 16);
            __builtin_prefetch(bptr0 + k0 + 256, 0, 0);  // global_prefetch next slab

            const int j0 = kk * 32;
#pragma unroll
            for (int mi = 0; mi < 4; ++mi) {
                FragU a;
                a.q[0] = *(const U4*)(arows[mi] + j0);
                a.q[1] = *(const U4*)(arows[mi] + j0 + 16);
                acc[mi][0] = __builtin_amdgcn_wmma_f32_16x16x32_bf16(
                    false, a.v, false, fb0.v, (short)0, acc[mi][0], false, false);
                acc[mi][1] = __builtin_amdgcn_wmma_f32_16x16x32_bf16(
                    false, a.v, false, fb1.v, (short)0, acc[mi][1], false, false);
            }
        }
        // fold per-expert result with gate weight w[b,n]
#pragma unroll
        for (int mi = 0; mi < 4; ++mi)
#pragma unroll
            for (int r = 0; r < 8; ++r) {
                float wv = s_w[(mrow + mi * 16 + r + ksel * 8) * 16 + n];
                outa[mi][0][r] += wv * acc[mi][0][r];
                outa[mi][1][r] += wv * acc[mi][1][r];
            }
    }

#pragma unroll
    for (int mi = 0; mi < 4; ++mi)
#pragma unroll
        for (int ni = 0; ni < 2; ++ni)
#pragma unroll
            for (int r = 0; r < 8; ++r) {
                const int grow = mblock + mrow + mi * 16 + r + ksel * 8;
                const int gcol = ncol + ni * 16 + arow;
                out[(size_t)grow * DS + gcol] = outa[mi][ni][r];
            }
}

// ---------------------------------------------------------------------------
extern "C" void kernel_launch(void* const* d_in, const int* in_sizes, int n_in,
                              void* d_out, int out_size, void* d_ws, size_t ws_size,
                              hipStream_t stream) {
    const float* x    = (const float*)d_in[0];
    const float* xtar = (const float*)d_in[1];
    const float* BA   = (const float*)d_in[2];
    const float* CA   = (const float*)d_in[3];
    const float* W1   = (const float*)d_in[4];
    const float* b1   = (const float*)d_in[5];
    const float* W2   = (const float*)d_in[6];
    const float* b2   = (const float*)d_in[7];
    const float* W3   = (const float*)d_in[8];
    const float* b3   = (const float*)d_in[9];
    const float* W4   = (const float*)d_in[10];
    const float* b4   = (const float*)d_in[11];
    float* out = (float*)d_out;

    char* ws = (char*)d_ws;
    unsigned short* diff_bf = (unsigned short*)(ws);                       // 16 MB
    unsigned short* afat    = (unsigned short*)(ws + 16777216);            // 2 MB
    unsigned short* w1bf    = (unsigned short*)(ws + 18874368);            // 64 KB
    float*          c1      = (float*)(ws + 18939904);                     // 512 B
    unsigned short* h1bf    = (unsigned short*)(ws + 18940416);            // 8 MB
    float*          wgate   = (float*)(ws + 27329024);                     // 2 MB

    prep_diff_kernel<<<(NBATCH * DS) / (4 * 256), 256, 0, stream>>>(x, xtar, diff_bf);
    prep_afat_kernel<<<(DS * KTOT) / (4 * 256), 256, 0, stream>>>(BA, CA, afat);
    prep_w1c1_kernel<<<1, 128, 0, stream>>>(W1, b1, xtar, w1bf, c1);
    gemm_h1_kernel<<<NBATCH / 256, 256, 0, stream>>>(diff_bf, w1bf, c1, h1bf);
    gating_tail_kernel<<<NBATCH / 256, 256, 0, stream>>>(h1bf, W2, b2, W3, b3, W4, b4, wgate);
    dim3 grid(NBATCH / 256, DS / 64);
    moe_main_gemm_kernel<<<grid, 256, 0, stream>>>(diff_bf, afat, wgate, out);
}
